// SAXSMSAAttention_40570261078771
// MI455X (gfx1250) — matrix operand order repaired and checked by reading, not compiled, not run on hardware
//
#include <hip/hip_runtime.h>
#include <hip/hip_bf16.h>

typedef __attribute__((ext_vector_type(16))) _Float16 v16h;
typedef __attribute__((ext_vector_type(8)))  float    v8f;

// -------------------------------------------------------------------------
// Kernel 1: delta[c] = sum_d (ln_k_b[0]*Wv[0,d] + bv[d]) * Wo[d,c] + bo[c]
// One wave32, WMMA f32_16x16x32_f16. A-matrix rows are all broadcast copies
// of v_vec, so every row of D equals the desired 16-wide delta block.
// -------------------------------------------------------------------------
__global__ __launch_bounds__(32)
void saxs_delta_wmma_kernel(const float* __restrict__ ln_k_b,
                            const float* __restrict__ Wv,
                            const float* __restrict__ bv,
                            const float* __restrict__ Wo,
                            const float* __restrict__ bo,
                            float* __restrict__ delta) {
  const int lane = threadIdx.x;            // 0..31
  const float lkb = ln_k_b[0];

  // A fragment (16-bit A 16x32 layout): element e of lane group g holds
  //   K = 32*kb + (e<8 ? e : e+8) + (g ? 8 : 0)
  const int abase = (lane >= 16) ? 8 : 0;
  // B fragment (16-bit B 32x16 layout): lane n holds column N = n%16,
  //   element e holds K = 32*kb + (e<8 ? e : e+8) + (lane>=16 ? 8 : 0)
  const int bN    = lane & 15;
  const int bbase = (lane >= 16) ? 8 : 0;

  for (int cb = 0; cb < 16; ++cb) {        // 16 column blocks of 16
    v8f acc = {};
    for (int kb = 0; kb < 8; ++kb) {       // K = 256 in steps of 32
      v16h a, b;
#pragma unroll
      for (int e = 0; e < 16; ++e) {
        const int ka = 32 * kb + ((e < 8) ? e : e + 8) + abase;
        a[e] = (_Float16)(lkb * Wv[ka] + bv[ka]);           // broadcast v_vec
        const int kw = 32 * kb + ((e < 8) ? e : e + 8) + bbase;
        b[e] = (_Float16)Wo[kw * 256 + cb * 16 + bN];
      }
      acc = __builtin_amdgcn_wmma_f32_16x16x32_f16(
          /*neg_a=*/false, a, /*neg_b=*/false, b,
          /*c_mod=*/(short)0, acc, /*reuse_a=*/false, /*reuse_b=*/false);
    }
    // Every D row is identical; row M=0 lives in acc[0] on lanes 0..15.
    if (lane < 16) {
      const int c = cb * 16 + lane;
      delta[c] = acc[0] + bo[c];
    }
  }
}

// -------------------------------------------------------------------------
// Kernel 2: out[i] = msa[i] + delta[i mod 256], float4-vectorized stream.
// 32 MB total traffic -> HBM-bound (~1.4 us at 23.3 TB/s).
// -------------------------------------------------------------------------
__global__ __launch_bounds__(256)
void saxs_broadcast_add_kernel(const float4* __restrict__ msa,
                               const float* __restrict__ delta,
                               float4* __restrict__ out, int n4) {
  __shared__ float4 sdelta[64];            // 256 floats = 64 float4
  if (threadIdx.x < 64) sdelta[threadIdx.x] = ((const float4*)delta)[threadIdx.x];
  __syncthreads();

  const int stride = gridDim.x * blockDim.x;
  for (int i = blockIdx.x * blockDim.x + threadIdx.x; i < n4; i += stride) {
    float4 m = msa[i];
    float4 d = sdelta[i & 63];             // (4*i) mod 256, as float4 index
    out[i] = make_float4(m.x + d.x, m.y + d.y, m.z + d.z, m.w + d.w);
  }
}

extern "C" void kernel_launch(void* const* d_in, const int* in_sizes, int n_in,
                              void* d_out, int out_size, void* d_ws, size_t ws_size,
                              hipStream_t stream) {
  // setup_inputs order:
  // 0 msa, 1 saxs, 2 ln_q_w, 3 ln_q_b, 4 ln_k_w, 5 ln_k_b,
  // 6 Wq, 7 bq, 8 Wk, 9 bk, 10 Wv, 11 bv, 12 Wo, 13 bo
  const float* msa    = (const float*)d_in[0];
  const float* ln_k_b = (const float*)d_in[5];
  const float* Wv     = (const float*)d_in[10];
  const float* bv     = (const float*)d_in[11];
  const float* Wo     = (const float*)d_in[12];
  const float* bo     = (const float*)d_in[13];

  float* delta = (float*)d_ws;             // 256 floats of scratch
  float* out   = (float*)d_out;

  saxs_delta_wmma_kernel<<<1, 32, 0, stream>>>(ln_k_b, Wv, bv, Wo, bo, delta);

  const int n4 = out_size / 4;             // 4,194,304 floats -> 1,048,576 float4
  saxs_broadcast_add_kernel<<<2048, 256, 0, stream>>>(
      (const float4*)msa, delta, (float4*)out, n4);
}
